// MHSA_37434934952705
// MI455X (gfx1250) — matrix operand order repaired and checked by reading, not compiled, not run on hardware
//
#include <hip/hip_runtime.h>
#include <hip/hip_bf16.h>
#include <math.h>
#include <stdint.h>

#define BB 4
#define CC 2048
#define KK 128
#define HH 8

typedef float  v8f   __attribute__((ext_vector_type(8)));
typedef __bf16 v16bf __attribute__((ext_vector_type(16)));
typedef __bf16 bf16x8 __attribute__((ext_vector_type(8)));

static __device__ __forceinline__ v8f wmma_bf16(v16bf a, v16bf b, v8f c) {
  // D = A(16x32 bf16) * B(32x16 bf16) + C(16x16 f32)
  return __builtin_amdgcn_wmma_f32_16x16x32_bf16(false, a, false, b, (short)0, c, false, false);
}

// CDNA5 async direct global->LDS copy (ASYNCcnt-tracked), 16B per lane.
// ldsOff: LDS byte offset (low 32 bits of a generic pointer to __shared__).
static __device__ __forceinline__ void async_g2l_b128(unsigned ldsOff, const void* gaddr) {
  asm volatile("global_load_async_to_lds_b128 %0, %1, off"
               :: "v"(ldsOff), "v"((unsigned long long)(uintptr_t)gaddr)
               : "memory");
}

// ---------------------------------------------------------------------------
// K0a: column sums of Wq / Wk over the out-feature (j) axis per head.
// wqs[h][m] = sum_j Wq[h*K+j][m]  (same for wks)
// ---------------------------------------------------------------------------
__global__ void k_wsum(const float* __restrict__ Wq, const float* __restrict__ Wk,
                       float* __restrict__ wqs, float* __restrict__ wks) {
  int idx = blockIdx.x * blockDim.x + threadIdx.x;   // 0..2047
  int sel = idx >> 10;                               // 0: Wq, 1: Wk
  int rem = idx & 1023;                              // h*128 + m
  int h = rem >> 7, m = rem & 127;
  const float* W = sel ? Wk : Wq;
  float acc = 0.f;
  for (int j = 0; j < KK; ++j) acc += W[(h * KK + j) * KK + m];
  if (sel) wks[rem] = acc; else wqs[rem] = acc;
}

// ---------------------------------------------------------------------------
// K0b: fused value/output weights: M_h = Wo_h @ Wv_h  (per head 128x128), bf16.
// mt2[h][kp][m] = sum_e Wo[kp][h*K+e] * Wv[h*K+e][m]
// ---------------------------------------------------------------------------
__global__ void k_mt2(const float* __restrict__ Wv, const float* __restrict__ Wo,
                      __bf16* __restrict__ mt2) {
  int h  = blockIdx.x >> 7;
  int kp = blockIdx.x & 127;
  int m  = threadIdx.x;
  const float* wo = Wo + (size_t)kp * (HH * KK) + h * KK;
  const float* wv = Wv + (size_t)(h * KK) * KK + m;
  float acc = 0.f;
  for (int e = 0; e < KK; ++e) acc = fmaf(wo[e], wv[(size_t)e * KK], acc);
  mt2[((size_t)h * KK + kp) * KK + m] = (__bf16)acc;
}

// ---------------------------------------------------------------------------
// K1: a[b][h][t] = x_row . wqs[h];  s[b][h][i] = (x_row . wks[h]) / sqrt(K)
// ---------------------------------------------------------------------------
__global__ void k_qk(const float* __restrict__ x, const float* __restrict__ wqs,
                     const float* __restrict__ wks, float* __restrict__ aArr,
                     float* __restrict__ sArr) {
  int tid  = threadIdx.x;
  int row  = blockIdx.x * 16 + (tid >> 4);   // 0..B*C-1
  int slot = tid & 15;                       // 0..7 -> q, 8..15 -> k
  int b = row / CC, t = row % CC;
  int h = slot & 7;
  const float* xr = x + (size_t)row * KK;
  const float* w  = (slot < 8 ? wqs : wks) + h * KK;
  float acc = 0.f;
  for (int m = 0; m < KK; ++m) acc = fmaf(xr[m], w[m], acc);
  if (slot < 8) aArr[((size_t)b * HH + h) * CC + t] = acc;
  else          sArr[((size_t)b * HH + h) * CC + t] = acc * 0.08838834764831845f; // 1/sqrt(128)
}

// ---------------------------------------------------------------------------
// K1b: per (b,h): smin/smax over i, then per t the closed-form row max
//      m_t = max(a_t*smax, a_t*smin) and zinv_t = 1 / sum_i exp(a_t*s_i - m_t)
// ---------------------------------------------------------------------------
__global__ void k_mz(const float* __restrict__ aArr, const float* __restrict__ sArr,
                     float* __restrict__ mArr, float* __restrict__ zArr) {
  __shared__ float sSh[CC];
  __shared__ float rmin[256];
  __shared__ float rmax[256];
  int bh = blockIdx.x;
  int tid = threadIdx.x;
  const float* sp = sArr + (size_t)bh * CC;
  float mn = 1e30f, mx = -1e30f;
  for (int i = tid; i < CC; i += 256) {
    float v = sp[i];
    sSh[i] = v;
    mn = fminf(mn, v);
    mx = fmaxf(mx, v);
  }
  rmin[tid] = mn; rmax[tid] = mx;
  __syncthreads();
  for (int st = 128; st > 0; st >>= 1) {
    if (tid < st) {
      rmin[tid] = fminf(rmin[tid], rmin[tid + st]);
      rmax[tid] = fmaxf(rmax[tid], rmax[tid + st]);
    }
    __syncthreads();
  }
  float smin = rmin[0], smax = rmax[0];
  for (int t = tid; t < CC; t += 256) {
    float aval = aArr[(size_t)bh * CC + t];
    float mv = fmaxf(aval * smax, aval * smin);
    float z = 0.f;
    for (int i = 0; i < CC; ++i) z += __expf(fmaf(aval, sSh[i], -mv));
    mArr[(size_t)bh * CC + t] = mv;
    zArr[(size_t)bh * CC + t] = 1.0f / z;
  }
}

// ---------------------------------------------------------------------------
// K2: U GEMM via WMMA bf16:  ut[b][h][kp][i] = sum_m x[b][i][m] * M_h[kp][m]
// Stored transposed (kp-major, i contiguous) so K3's B staging is a flat copy.
// Block = 128 i-rows (8 waves x 16), full 128 kp columns (8 n-tiles / wave).
// ---------------------------------------------------------------------------
__global__ void k_u(const float* __restrict__ x, const __bf16* __restrict__ mt2,
                    __bf16* __restrict__ ut) {
  int iTile = blockIdx.x;               // 0..15
  int h = blockIdx.y, b = blockIdx.z;
  int wave = threadIdx.x >> 5, lane = threadIdx.x & 31;
  int Lm = lane & 15, Lhi = lane >> 4;
  int r0 = iTile * 128 + wave * 16;     // i-row base for this wave
  const float*  xrow = x   + ((size_t)b * CC + r0 + Lm) * KK;
  const __bf16* mtH  = mt2 + (size_t)h * KK * KK;

  v8f acc[8] = {};
  #pragma unroll
  for (int k0 = 0; k0 < KK; k0 += 32) {
    // A fragment (x, f32 -> bf16): element e<8 -> K = k0+8*Lhi+e, e>=8 -> +16
    v16bf af;
    const float* xa = xrow + k0 + 8 * Lhi;
    const float* xb = xrow + k0 + 16 + 8 * Lhi;
    #pragma unroll
    for (int e = 0; e < 8; ++e) {
      af[e]     = (__bf16)xa[e];
      af[8 + e] = (__bf16)xb[e];
    }
    #pragma unroll
    for (int n = 0; n < 8; ++n) {
      // B fragment: B[K=m][N=kp] = M_h[kp][m]; mt2 is [kp][m] (K contiguous)
      const __bf16* br = mtH + ((size_t)(n * 16 + Lm)) * KK + k0 + 8 * Lhi;
      bf16x8 b0 = *(const bf16x8*)br;
      bf16x8 b1 = *(const bf16x8*)(br + 16);
      v16bf bf;
      #pragma unroll
      for (int e = 0; e < 8; ++e) { bf[e] = b0[e]; bf[8 + e] = b1[e]; }
      acc[n] = wmma_bf16(af, bf, acc[n]);
    }
  }
  // D layout: VGPR d, lanes 0-15 -> M=d, lanes 16-31 -> M=d+8; N = lane%16
  #pragma unroll
  for (int n = 0; n < 8; ++n) {
    int kp = n * 16 + Lm;
    __bf16* dst = ut + (((size_t)b * HH + h) * KK + kp) * CC;
    #pragma unroll
    for (int d = 0; d < 8; ++d) {
      int i = r0 + d + 8 * Lhi;
      dst[i] = (__bf16)acc[n][d];
    }
  }
}

// ---------------------------------------------------------------------------
// K3: fused attention + output projection.
// Block = (b, 128 t-rows). Each wave owns 16 t-rows, 8 f32 accum tiles
// covering all 128 output columns. Per head: P tile built in registers
// (exp of rank-1 logits, pre-normalized by zinv -> accumulate across heads
// with no rescaling). U chunks are staged with CDNA5 async global->LDS
// copies (ASYNCcnt) into a double buffer in [N][K] order.
// ---------------------------------------------------------------------------
__global__ void k_attn(const float* __restrict__ aArr, const float* __restrict__ sArr,
                       const float* __restrict__ mArr, const float* __restrict__ zArr,
                       const __bf16* __restrict__ ut, const float* __restrict__ bo,
                       float* __restrict__ y) {
  __shared__ float sSh[CC];                              // 8 KB: s row for this head
  __shared__ __align__(16) __bf16 uSh[2][128 * 32];      // 2 x 8 KB: U chunks, [kp][j]
  int tTile = blockIdx.x, b = blockIdx.y;
  int tid = threadIdx.x;
  int wave = tid >> 5, lane = tid & 31;
  int Lm = lane & 15, Lhi = lane >> 4;
  int t0 = tTile * 128 + wave * 16;

  v8f acc[8] = {};
  int ukp   = tid >> 1;          // 0..127: kp row this thread stages
  int uhalf = (tid & 1) * 16;    // 0 or 16: which 16 j's
  // LDS byte offsets of this thread's staging slots (generic ptr low 32 bits)
  unsigned ldsOff0 = (unsigned)(uintptr_t)&uSh[0][ukp * 32 + uhalf];
  unsigned ldsOff1 = (unsigned)(uintptr_t)&uSh[1][ukp * 32 + uhalf];

  for (int h = 0; h < HH; ++h) {
    size_t bh = (size_t)b * HH + h;
    __syncthreads();                                   // sSh readers from prev h done
    for (int i = tid; i < CC; i += 256) sSh[i] = sArr[bh * CC + i];
    __syncthreads();
    float aval = aArr[bh * CC + t0 + Lm];
    float mval = mArr[bh * CC + t0 + Lm];
    float zval = zArr[bh * CC + t0 + Lm];
    const __bf16* uSrc = ut + (bh * KK + ukp) * CC + uhalf;  // this thread's row

    // Prologue: async-stage chunk 0 into buffer 0
    async_g2l_b128(ldsOff0, uSrc);
    async_g2l_b128(ldsOff0 + 16, uSrc + 8);

    for (int ci = 0; ci < CC / 32; ++ci) {
      int buf = ci & 1;
      if (ci + 1 < CC / 32) {
        // Issue next chunk into the other buffer, then wait for the older one.
        async_g2l_b128(buf ? ldsOff0 : ldsOff1, uSrc + (ci + 1) * 32);
        async_g2l_b128((buf ? ldsOff0 : ldsOff1) + 16, uSrc + (ci + 1) * 32 + 8);
        asm volatile("s_wait_asynccnt 0x2" ::: "memory");
      } else {
        asm volatile("s_wait_asynccnt 0x0" ::: "memory");
      }
      __syncthreads();                                 // chunk ci resident for all waves
      const __bf16* uCur = &uSh[buf][0];
      int i0 = ci * 32;

      // A fragment: normalized probabilities P[t, i0+j], computed in registers
      v16bf af;
      const float* sa = &sSh[i0 + 8 * Lhi];
      const float* sb = &sSh[i0 + 16 + 8 * Lhi];
      #pragma unroll
      for (int e = 0; e < 8; ++e) {
        af[e]     = (__bf16)(__expf(fmaf(aval, sa[e], -mval)) * zval);
        af[8 + e] = (__bf16)(__expf(fmaf(aval, sb[e], -mval)) * zval);
      }
      #pragma unroll
      for (int n = 0; n < 8; ++n) {
        const __bf16* br = uCur + (n * 16 + Lm) * 32 + 8 * Lhi;
        bf16x8 b0 = *(const bf16x8*)br;
        bf16x8 b1 = *(const bf16x8*)(br + 16);
        v16bf bf;
        #pragma unroll
        for (int e = 0; e < 8; ++e) { bf[e] = b0[e]; bf[8 + e] = b1[e]; }
        acc[n] = wmma_bf16(af, bf, acc[n]);
      }
      __syncthreads();  // all waves done with buf before it is re-issued (ci+2)
    }
  }

  // Epilogue: add bias, store y[b][t][kp] (f32)
  #pragma unroll
  for (int n = 0; n < 8; ++n) {
    int kp = n * 16 + Lm;
    float bias = bo[kp];
    #pragma unroll
    for (int d = 0; d < 8; ++d) {
      int t = t0 + d + 8 * Lhi;
      y[((size_t)b * CC + t) * KK + kp] = acc[n][d] + bias;
    }
  }
}

// ---------------------------------------------------------------------------
extern "C" void kernel_launch(void* const* d_in, const int* in_sizes, int n_in,
                              void* d_out, int out_size, void* d_ws, size_t ws_size,
                              hipStream_t stream) {
  const float* x  = (const float*)d_in[0];
  const float* Wq = (const float*)d_in[1];
  const float* Wk = (const float*)d_in[2];
  const float* Wv = (const float*)d_in[3];
  const float* Wo = (const float*)d_in[4];
  const float* bo = (const float*)d_in[5];
  float* y = (float*)d_out;

  char* p = (char*)d_ws;
  auto alloc = [&](size_t bytes) -> char* {
    char* r = p;
    p += (bytes + 255) & ~(size_t)255;
    return r;
  };
  float*  wqs  = (float*)alloc((size_t)HH * KK * 4);
  float*  wks  = (float*)alloc((size_t)HH * KK * 4);
  float*  aArr = (float*)alloc((size_t)BB * HH * CC * 4);
  float*  sArr = (float*)alloc((size_t)BB * HH * CC * 4);
  float*  mArr = (float*)alloc((size_t)BB * HH * CC * 4);
  float*  zArr = (float*)alloc((size_t)BB * HH * CC * 4);
  __bf16* mt2  = (__bf16*)alloc((size_t)HH * KK * KK * 2);
  __bf16* ut   = (__bf16*)alloc((size_t)BB * HH * KK * CC * 2);   // ~16 MB

  k_wsum<<<8, 256, 0, stream>>>(Wq, Wk, wqs, wks);
  k_mt2 <<<HH * KK, KK, 0, stream>>>(Wv, Wo, mt2);
  k_qk  <<<BB * CC / 16, 256, 0, stream>>>(x, wqs, wks, aArr, sArr);
  k_mz  <<<BB * HH, 256, 0, stream>>>(aArr, sArr, mArr, zArr);
  k_u   <<<dim3(CC / 128, HH, BB), 256, 0, stream>>>(x, mt2, ut);
  k_attn<<<dim3(CC / 128, BB), 256, 0, stream>>>(aArr, sArr, mArr, zArr, ut, bo, y);
}
